// DGCNN_9706626090046
// MI455X (gfx1250) — matrix operand order repaired and checked by reading, not compiled, not run on hardware
//
#include <hip/hip_runtime.h>

// DGCNN forward for MI455X (gfx1250), wave32 + WMMA f16.
// All GEMM operands pre-converted to padded f16 so WMMA fragments are
// contiguous 16B vector loads (global_load_b128 / ds_load_b128), no guards.
// Workspace budget (~35 MB).

#define K_NN     20
#define NPTS     4096
#define BATCH    2
#define ROWCHUNK 512
#define SLOPE    0.2f
#define EPS_BN   1e-5f

typedef __attribute__((ext_vector_type(16))) _Float16 v16h;
typedef __attribute__((ext_vector_type(8)))  _Float16 v8h;
typedef __attribute__((ext_vector_type(8)))  float    v8f;

union AF { v16h v; v8h q[2]; _Float16 h[16]; };
union CF { v8f  v; float f[8]; };

#define WMMA_F16(A, B, C) \
  __builtin_amdgcn_wmma_f32_16x16x32_f16(false, (A), false, (B), (short)0, (C), false, false)

__device__ __forceinline__ float leaky(float y) { return y > 0.f ? y : SLOPE * y; }

// CDNA5 16-bit A 16x32 fragment: lane L (row M=L&15) holds K {b..b+7, b+16..b+23}, b = (L>=16)?8:0
// CDNA5 16-bit B 32x16 fragment: lane L (col N=L&15) holds K {b..b+7, b+16..b+23}, b = (L>=16)?8:0

// ---------------------------------------------------------------- utilities
__global__ void k_zero(float* p, int n) {
  int t = blockIdx.x * blockDim.x + threadIdx.x;
  if (t < n) p[t] = 0.f;
}

// f32 (rows x strideIn, C valid cols) -> f16 (rows x strideOut, zero padded)
__global__ void k_f2h_pad(const float* __restrict__ fin, int strideIn, int C,
                          _Float16* __restrict__ out, int strideOut, int rows) {
  int t = blockIdx.x * blockDim.x + threadIdx.x;
  if (t >= rows * strideOut) return;
  int r = t / strideOut, c = t - r * strideOut;
  out[t] = (c < C) ? (_Float16)fin[(long)r * strideIn + c] : (_Float16)0.f;
}

// x (B,3,N) -> xb (B,N,3)
__global__ void k_transpose(const float* __restrict__ x, float* __restrict__ xb) {
  int t = blockIdx.x * blockDim.x + threadIdx.x;
  int total = BATCH * NPTS * 3;
  if (t >= total) return;
  int b = t / (NPTS * 3);
  int r = t - b * NPTS * 3;
  int n = r / 3, c = r - n * 3;
  xb[t] = x[((long)b * 3 + c) * NPTS + n];
}

// xx[b*N+n] = sum_c feat[b,n,c]^2
__global__ void k_rownorm(const float* __restrict__ fin, float* __restrict__ xx,
                          int C, int stride) {
  int t = blockIdx.x * blockDim.x + threadIdx.x;
  if (t >= BATCH * NPTS) return;
  const float* row = fin + (long)t * stride;
  float s = 0.f;
  for (int c = 0; c < C; ++c) s += row[c] * row[c];
  xx[t] = s;
}

// ---------------------------------------------------------------- KNN: dist
// neg_d rows [i0, i0+ROWCHUNK) x all 4096 cols, one batch, from padded f16 features.
__global__ void k_dist(const _Float16* __restrict__ fH, const float* __restrict__ xxb,
                       float* __restrict__ dist, int padC, int i0, int nTiles) {
  int wave = blockIdx.x * (blockDim.x >> 5) + (threadIdx.x >> 5);
  int lane = threadIdx.x & 31;
  if (wave >= nTiles) return;                 // wave-uniform
  const int tilesJ = NPTS >> 4;
  int ti = wave / tilesJ, tj = wave - ti * tilesJ;
  int ibase = i0 + ti * 16, jbase = tj * 16;

  int arow = ibase + (lane & 15);
  int bcol = jbase + (lane & 15);
  const _Float16* ar = fH + (long)arow * padC + ((lane >= 16) ? 8 : 0);
  const _Float16* br = fH + (long)bcol * padC + ((lane >= 16) ? 8 : 0);

  CF acc;
#pragma unroll
  for (int r = 0; r < 8; ++r) acc.f[r] = 0.f;

  for (int kc0 = 0; kc0 < padC; kc0 += 32) {
    AF a, bm;
    a.q[0]  = *(const v8h*)(ar + kc0);
    a.q[1]  = *(const v8h*)(ar + kc0 + 16);
    bm.q[0] = *(const v8h*)(br + kc0);
    bm.q[1] = *(const v8h*)(br + kc0 + 16);
    acc.v = WMMA_F16(a.v, bm.v, acc.v);
  }

  float xxj = xxb[bcol];
#pragma unroll
  for (int r = 0; r < 8; ++r) {
    int irow = ibase + r + ((lane >= 16) ? 8 : 0);
    dist[(long)(irow - i0) * NPTS + jbase + (lane & 15)] = 2.f * acc.f[r] - xxb[irow] - xxj;
  }
}

// ---------------------------------------------------------------- KNN: top-k
__global__ void k_topk(const float* __restrict__ dist, int* __restrict__ idxb, int i0) {
  int t = blockIdx.x * blockDim.x + threadIdx.x;
  if (t >= ROWCHUNK) return;
  const float* row = dist + (long)t * NPTS;
  float bv[K_NN];
  int   bi[K_NN];
#pragma unroll
  for (int k = 0; k < K_NN; ++k) { bv[k] = -3.4e38f; bi[k] = 0; }
  for (int j = 0; j < NPTS; ++j) {
    float v = row[j];
    if (v > bv[K_NN - 1]) {
      int p = K_NN - 1;
      while (p > 0 && bv[p - 1] < v) { bv[p] = bv[p - 1]; bi[p] = bi[p - 1]; --p; }
      bv[p] = v; bi[p] = j;
    }
  }
  int* orow = idxb + (long)(i0 + t) * K_NN;
#pragma unroll
  for (int k = 0; k < K_NN; ++k) orow[k] = bi[k];
}

// ---------------------------------------------------------------- edge conv
// 32x16 h tile (rows = 20 edge rows of one point padded to 32, cols = 16 out ch)
// from LDS-staged f16 edge features and padded f16 weights.
__device__ __forceinline__ void ec_htile(const _Float16* __restrict__ sf,
                                         const _Float16* __restrict__ WhP,
                                         int padC2, int o0, int lane,
                                         CF& h0, CF& h1) {
#pragma unroll
  for (int r = 0; r < 8; ++r) { h0.f[r] = 0.f; h1.f[r] = 0.f; }
  int r = lane & 15;
  const _Float16* s0 = sf + r * padC2 + ((lane >= 16) ? 8 : 0);
  const _Float16* s1 = s0 + 16 * padC2;
  const _Float16* wp = WhP + (long)(o0 + r) * padC2 + ((lane >= 16) ? 8 : 0);
  for (int kc0 = 0; kc0 < padC2; kc0 += 32) {
    AF a0, a1, bm;
    a0.q[0] = *(const v8h*)(s0 + kc0);
    a0.q[1] = *(const v8h*)(s0 + kc0 + 16);
    a1.q[0] = *(const v8h*)(s1 + kc0);
    a1.q[1] = *(const v8h*)(s1 + kc0 + 16);
    bm.q[0] = *(const v8h*)(wp + kc0);
    bm.q[1] = *(const v8h*)(wp + kc0 + 16);
    h0.v = WMMA_F16(a0.v, bm.v, h0.v);
    h1.v = WMMA_F16(a1.v, bm.v, h1.v);
  }
}

// Stage [nbr - center, center] rows (k=0..19; 20..31 zero) into LDS as f16.
__device__ __forceinline__ void ec_stage(_Float16* sf, const float* __restrict__ fin,
                                         const int* __restrict__ idxAll,
                                         int Cin, int inStride, int b, int n, int lane,
                                         int padC2) {
  const int twoC = 2 * Cin;
  const float* fb  = fin + (long)b * NPTS * inStride;
  const float* ctr = fb + (long)n * inStride;
  _Float16* row = sf + lane * padC2;
  if (lane < K_NN) {
    int j = idxAll[((long)b * NPTS + n) * K_NN + lane];
    const float* nbr = fb + (long)j * inStride;
    for (int c = 0; c < Cin; ++c) {
      row[c]       = (_Float16)(nbr[c] - ctr[c]);
      row[Cin + c] = (_Float16)ctr[c];
    }
    for (int c = twoC; c < padC2; ++c) row[c] = (_Float16)0.f;
  } else {
    for (int c = 0; c < padC2; ++c) row[c] = (_Float16)0.f;
  }
  __syncthreads();
}

// Pass 1: per-output-channel sum / sumsq of pre-BN h over all (b,n,k).
__global__ void k_ec_stats(const float* __restrict__ fin, int Cin, int inStride,
                           const int* __restrict__ idxAll, const _Float16* __restrict__ WhP,
                           int Cout, float* __restrict__ sum, float* __restrict__ sumsq) {
  extern __shared__ _Float16 sf[];
  const int padC2 = (2 * Cin + 31) & ~31;
  int bn = blockIdx.x;
  int b = bn / NPTS, n = bn - b * NPTS;
  int lane = threadIdx.x;
  ec_stage(sf, fin, idxAll, Cin, inStride, b, n, lane, padC2);

  for (int ot = 0; ot < (Cout >> 4); ++ot) {
    CF h0, h1;
    ec_htile(sf, WhP, padC2, ot * 16, lane, h0, h1);
    float s = 0.f, s2 = 0.f;
#pragma unroll
    for (int r = 0; r < 8; ++r) { float v = h0.f[r]; s += v; s2 += v * v; }
    if (lane < 16) {
#pragma unroll
      for (int r = 0; r < 4; ++r) { float v = h1.f[r]; s += v; s2 += v * v; }
    }
    int o = ot * 16 + (lane & 15);
    atomicAdd(&sum[o], s);
    atomicAdd(&sumsq[o], s2);
  }
}

// Pass 2: recompute h, apply BN scale/shift + leaky, max over K, write feats slice.
__global__ void k_ec_out(const float* __restrict__ fin, int Cin, int inStride,
                         const int* __restrict__ idxAll, const _Float16* __restrict__ WhP,
                         int Cout, const float* __restrict__ scale,
                         const float* __restrict__ shift,
                         float* __restrict__ foutBase) {
  extern __shared__ _Float16 sf[];
  const int padC2 = (2 * Cin + 31) & ~31;
  int bn = blockIdx.x;
  int b = bn / NPTS, n = bn - b * NPTS;
  int lane = threadIdx.x;
  ec_stage(sf, fin, idxAll, Cin, inStride, b, n, lane, padC2);

  for (int ot = 0; ot < (Cout >> 4); ++ot) {
    CF h0, h1;
    ec_htile(sf, WhP, padC2, ot * 16, lane, h0, h1);
    int o = ot * 16 + (lane & 15);
    float sc = scale[o], sh = shift[o];
    float m = -3.4e38f;
#pragma unroll
    for (int r = 0; r < 8; ++r) m = fmaxf(m, leaky(h0.f[r] * sc + sh));
    if (lane < 16) {
#pragma unroll
      for (int r = 0; r < 4; ++r) m = fmaxf(m, leaky(h1.f[r] * sc + sh));
    }
    float other = __shfl_xor(m, 16, 32);   // pair lanes holding same column
    m = fmaxf(m, other);
    if (lane < 16)
      foutBase[(long)bn * 512 + o] = m;    // feats stride is always 512
  }
}

// ---------------------------------------------------------------- BN params
__global__ void k_bnparams(const float* __restrict__ sum, const float* __restrict__ sumsq,
                           const float* __restrict__ g, const float* __restrict__ bb,
                           float invR, float* __restrict__ scale, float* __restrict__ shift,
                           int C) {
  int o = blockIdx.x * blockDim.x + threadIdx.x;
  if (o >= C) return;
  float m = sum[o] * invR;
  float v = sumsq[o] * invR - m * m;
  float sc = g[o] * rsqrtf(v + EPS_BN);
  scale[o] = sc;
  shift[o] = bb[o] - m * sc;
}

// ---------------------------------------------------------------- final FC
// h = cat(8192,512) x W5^T(512,1024), all-f16 operands.
__global__ void k_fc_stats(const _Float16* __restrict__ fH, const _Float16* __restrict__ W5P,
                           float* __restrict__ sum, float* __restrict__ sumsq, int nTiles) {
  int wave = blockIdx.x * (blockDim.x >> 5) + (threadIdx.x >> 5);
  int lane = threadIdx.x & 31;
  if (wave >= nTiles) return;
  const int otiles = 1024 >> 4;
  int tm = wave / otiles, ot = wave - tm * otiles;
  int m0 = tm * 16, o0 = ot * 16;
  int r = lane & 15;
  const _Float16* ar = fH + (long)(m0 + r) * 512 + ((lane >= 16) ? 8 : 0);
  const _Float16* wp = W5P + (long)(o0 + r) * 512 + ((lane >= 16) ? 8 : 0);
  CF acc;
#pragma unroll
  for (int rr = 0; rr < 8; ++rr) acc.f[rr] = 0.f;
  for (int kc0 = 0; kc0 < 512; kc0 += 32) {
    __builtin_prefetch(ar + kc0 + 64, 0, 1);
    AF a, bm;
    a.q[0]  = *(const v8h*)(ar + kc0);
    a.q[1]  = *(const v8h*)(ar + kc0 + 16);
    bm.q[0] = *(const v8h*)(wp + kc0);
    bm.q[1] = *(const v8h*)(wp + kc0 + 16);
    acc.v = WMMA_F16(a.v, bm.v, acc.v);
  }
  float s = 0.f, s2 = 0.f;
#pragma unroll
  for (int rr = 0; rr < 8; ++rr) { float v = acc.f[rr]; s += v; s2 += v * v; }
  int o = o0 + (lane & 15);
  atomicAdd(&sum[o], s);
  atomicAdd(&sumsq[o], s2);
}

__global__ void k_fc_out(const _Float16* __restrict__ fH, const _Float16* __restrict__ W5P,
                         const float* __restrict__ scale, const float* __restrict__ shift,
                         float* __restrict__ out, int nTiles) {
  int wave = blockIdx.x * (blockDim.x >> 5) + (threadIdx.x >> 5);
  int lane = threadIdx.x & 31;
  if (wave >= nTiles) return;
  const int otiles = 1024 >> 4;
  int tm = wave / otiles, ot = wave - tm * otiles;
  int m0 = tm * 16, o0 = ot * 16;
  int r = lane & 15;
  const _Float16* ar = fH + (long)(m0 + r) * 512 + ((lane >= 16) ? 8 : 0);
  const _Float16* wp = W5P + (long)(o0 + r) * 512 + ((lane >= 16) ? 8 : 0);
  CF acc;
#pragma unroll
  for (int rr = 0; rr < 8; ++rr) acc.f[rr] = 0.f;
  for (int kc0 = 0; kc0 < 512; kc0 += 32) {
    __builtin_prefetch(ar + kc0 + 64, 0, 1);
    AF a, bm;
    a.q[0]  = *(const v8h*)(ar + kc0);
    a.q[1]  = *(const v8h*)(ar + kc0 + 16);
    bm.q[0] = *(const v8h*)(wp + kc0);
    bm.q[1] = *(const v8h*)(wp + kc0 + 16);
    acc.v = WMMA_F16(a.v, bm.v, acc.v);
  }
  int o = o0 + (lane & 15);
  float sc = scale[o], sh = shift[o];
#pragma unroll
  for (int rr = 0; rr < 8; ++rr) {
    int mrow = m0 + rr + ((lane >= 16) ? 8 : 0);
    int b = mrow >> 12, n = mrow & (NPTS - 1);
    out[((long)b * 1024 + o) * NPTS + n] = leaky(acc.f[rr] * sc + sh);
  }
}

// ================================================================ launcher
extern "C" void kernel_launch(void* const* d_in, const int* in_sizes, int n_in,
                              void* d_out, int out_size, void* d_ws, size_t ws_size,
                              hipStream_t stream) {
  const float* x = (const float*)d_in[0];
  const float* W[5]  = {(const float*)d_in[1], (const float*)d_in[4], (const float*)d_in[7],
                        (const float*)d_in[10], (const float*)d_in[13]};
  const float* g[5]  = {(const float*)d_in[2], (const float*)d_in[5], (const float*)d_in[8],
                        (const float*)d_in[11], (const float*)d_in[14]};
  const float* bb[5] = {(const float*)d_in[3], (const float*)d_in[6], (const float*)d_in[9],
                        (const float*)d_in[12], (const float*)d_in[15]};
  float* out = (float*)d_out;

  // --- workspace carve (256B aligned) ---
  char* p = (char*)d_ws;
  auto carve = [&](size_t bytes) -> void* {
    void* r = (void*)p;
    p += (bytes + 255) & ~(size_t)255;
    return r;
  };
  float*     feats = (float*)carve((size_t)BATCH * NPTS * 512 * 4);   // x1|x2|x3|x4
  float*     xb    = (float*)carve((size_t)BATCH * NPTS * 3 * 4);
  float*     xx    = (float*)carve((size_t)BATCH * NPTS * 4);
  float*     dist  = (float*)carve((size_t)ROWCHUNK * NPTS * 4);
  float*     sum   = (float*)carve(1024 * 4);
  float*     sumsq = (float*)carve(1024 * 4);
  float*     scale = (float*)carve(1024 * 4);
  float*     shift = (float*)carve(1024 * 4);
  int*       idx   = (int*)carve((size_t)BATCH * NPTS * K_NN * 4);
  _Float16*  featH = (_Float16*)carve((size_t)BATCH * NPTS * 512 * 2); // padded f16 features
  // padded f16 weights: strides 32,128,128,256,512
  const int  Kdim[5] = {6, 128, 128, 256, 512};
  const int  padK[5] = {32, 128, 128, 256, 512};
  const int  Crow[5] = {64, 64, 128, 256, 1024};
  _Float16*  WhP[5];
  for (int l = 0; l < 5; ++l) WhP[l] = (_Float16*)carve((size_t)Crow[l] * padK[l] * 2);

  // --- prep: transpose + padded f16 weights ---
  k_transpose<<<(BATCH * NPTS * 3 + 255) / 256, 256, 0, stream>>>(x, xb);
  for (int l = 0; l < 5; ++l) {
    int nel = Crow[l] * padK[l];
    k_f2h_pad<<<(nel + 255) / 256, 256, 0, stream>>>(W[l], Kdim[l], Kdim[l],
                                                     WhP[l], padK[l], Crow[l]);
  }

  // --- edge conv layers ---
  const int Cin[4]      = {3, 64, 64, 128};
  const int Cout[4]     = {64, 64, 128, 256};
  const int inStride[4] = {3, 512, 512, 512};
  const int inOff[4]    = {0, 0, 64, 128};
  const int outOff[4]   = {0, 64, 128, 256};

  const int distTiles  = (ROWCHUNK / 16) * (NPTS / 16);      // 8192 waves
  const int distBlocks = distTiles / 8;                      // 256-thread blocks

  for (int l = 0; l < 4; ++l) {
    const float* fin = (l == 0) ? xb : (feats + inOff[l]);
    const int padC = (Cin[l] + 31) & ~31;
    // padded f16 copy of KNN input + row norms
    {
      int nel = BATCH * NPTS * padC;
      k_f2h_pad<<<(nel + 255) / 256, 256, 0, stream>>>(fin, inStride[l], Cin[l],
                                                       featH, padC, BATCH * NPTS);
    }
    k_rownorm<<<(BATCH * NPTS + 255) / 256, 256, 0, stream>>>(fin, xx, Cin[l], inStride[l]);
    // KNN, chunked rows
    for (int b = 0; b < BATCH; ++b) {
      const _Float16* fHb = featH + (long)b * NPTS * padC;
      const float*    xxb = xx + (long)b * NPTS;
      int*            idxb = idx + (long)b * NPTS * K_NN;
      for (int i0 = 0; i0 < NPTS; i0 += ROWCHUNK) {
        k_dist<<<distBlocks, 256, 0, stream>>>(fHb, xxb, dist, padC, i0, distTiles);
        k_topk<<<(ROWCHUNK + 127) / 128, 128, 0, stream>>>(dist, idxb, i0);
      }
    }
    // edge conv: stats -> bn params -> output (max over K)
    const int padC2 = (2 * Cin[l] + 31) & ~31;
    const size_t lds = (size_t)32 * padC2 * 2;
    k_zero<<<(1024 + 255) / 256, 256, 0, stream>>>(sum, 1024);
    k_zero<<<(1024 + 255) / 256, 256, 0, stream>>>(sumsq, 1024);
    k_ec_stats<<<BATCH * NPTS, 32, lds, stream>>>(fin, Cin[l], inStride[l], idx, WhP[l],
                                                  Cout[l], sum, sumsq);
    k_bnparams<<<(Cout[l] + 63) / 64, 64, 0, stream>>>(sum, sumsq, g[l], bb[l],
                                                       1.f / (float)(BATCH * NPTS * K_NN),
                                                       scale, shift, Cout[l]);
    k_ec_out<<<BATCH * NPTS, 32, lds, stream>>>(fin, Cin[l], inStride[l], idx, WhP[l],
                                                Cout[l], scale, shift, feats + outOff[l]);
  }

  // --- final FC (512 -> 1024) + BN + leaky, transposed output ---
  {
    int nel = BATCH * NPTS * 512;   // f16 copy of cat features
    k_f2h_pad<<<(nel + 255) / 256, 256, 0, stream>>>(feats, 512, 512, featH, 512, BATCH * NPTS);
  }
  const int fcTiles  = (BATCH * NPTS / 16) * (1024 / 16);    // 32768 waves
  const int fcBlocks = fcTiles / 8;
  k_zero<<<(1024 + 255) / 256, 256, 0, stream>>>(sum, 1024);
  k_zero<<<(1024 + 255) / 256, 256, 0, stream>>>(sumsq, 1024);
  k_fc_stats<<<fcBlocks, 256, 0, stream>>>(featH, WhP[4], sum, sumsq, fcTiles);
  k_bnparams<<<(1024 + 63) / 64, 64, 0, stream>>>(sum, sumsq, g[4], bb[4],
                                                  1.f / (float)(BATCH * NPTS),
                                                  scale, shift, 1024);
  k_fc_out<<<fcBlocks, 256, 0, stream>>>(featH, WhP[4], scale, shift, out, fcTiles);
}
